// Conv_pillar_caf_73169062855070
// MI455X (gfx1250) — matrix operand order repaired and checked by reading, not compiled, not run on hardware
//
#include <hip/hip_runtime.h>
#include <hip/hip_bf16.h>

typedef __attribute__((ext_vector_type(16))) _Float16 v16h;
typedef __attribute__((ext_vector_type(8)))  float    v8f;

#define NPTS   32768
#define NBATCH 8
#define NTOT   (NPTS * NBATCH)      // 262144 points
#define MIDC   64
#define OUTC   128
#define RES    64
#define NCELL  (NBATCH * RES * RES) // 32768 cells
#define EPSV   1e-4f

// ---------------------------------------------------------------------------
// helpers
// ---------------------------------------------------------------------------
__device__ __forceinline__ float wred16(float v) {
  // sum across the 16 lanes of each half-wave (lanes 0-15 / 16-31)
  v += __shfl_xor(v, 1, 32);
  v += __shfl_xor(v, 2, 32);
  v += __shfl_xor(v, 4, 32);
  v += __shfl_xor(v, 8, 32);
  return v;
}

// ---------------------------------------------------------------------------
// trivial utility kernels
// ---------------------------------------------------------------------------
__global__ void zero_kernel(float* p, int n) {
  int i = blockIdx.x * blockDim.x + threadIdx.x;
  if (i < n) p[i] = 0.f;
}

__global__ void cvt_f16_kernel(const float* __restrict__ s, _Float16* __restrict__ d, int n) {
  int i = blockIdx.x * blockDim.x + threadIdx.x;
  if (i < n) d[i] = (_Float16)s[i];
}

// att2a weight is (64 x 128); only the first 64 input columns multiply per-point
// data (the gmax half is folded into a per-batch constant elsewhere).
__global__ void cvt_att2a_kernel(const float* __restrict__ s, _Float16* __restrict__ d) {
  int i = blockIdx.x * blockDim.x + threadIdx.x;        // 4096
  if (i < 64 * 64) {
    int oc = i >> 6, c = i & 63;
    d[i] = (_Float16)s[oc * 128 + c];
  }
}

// conv weight OIHW (64,64,3,3) -> f16 [oc][tap][ic]  (K-dim = tap*64+ic)
__global__ void cvt_convw_kernel(const float* __restrict__ s, _Float16* __restrict__ d) {
  int i = blockIdx.x * blockDim.x + threadIdx.x;        // 36864
  if (i < 64 * 576) {
    int oc = i / 576, r = i % 576, tap = r / 64, ic = r % 64;
    d[i] = (_Float16)s[(oc * 64 + ic) * 9 + tap];
  }
}

// stat layout per layer (512 floats): [sum(128) | sumsq(128) | scale(128) | shift(128)]
__global__ void finalize_bn_kernel(float* stat, const float* __restrict__ g,
                                   const float* __restrict__ beta, int ch, float invn) {
  int c = threadIdx.x;
  if (c < ch) {
    float mean = stat[c] * invn;
    float var  = stat[128 + c] * invn - mean * mean;
    float sc   = g[c] * rsqrtf(var + 1e-4f);
    stat[256 + c] = sc;
    stat[384 + c] = beta[c] - mean * sc;
  }
}

// ---------------------------------------------------------------------------
// coordinate preprocessing
// ---------------------------------------------------------------------------
__global__ void batch_mean_kernel(const float* __restrict__ coords, float* __restrict__ meanb) {
  __shared__ float red[256];
  int b = blockIdx.x;
  for (int a = 0; a < 3; ++a) {
    const float* src = coords + ((size_t)b * 3 + a) * NPTS;
    float s = 0.f;
    for (int p = threadIdx.x; p < NPTS; p += blockDim.x) s += src[p];
    red[threadIdx.x] = s; __syncthreads();
    for (int off = 128; off > 0; off >>= 1) {
      if (threadIdx.x < off) red[threadIdx.x] += red[threadIdx.x + off];
      __syncthreads();
    }
    if (threadIdx.x == 0) meanb[b * 3 + a] = red[0] * (1.f / NPTS);
    __syncthreads();
  }
}

__global__ void batch_maxnorm_kernel(const float* __restrict__ coords,
                                     const float* __restrict__ meanb, float* __restrict__ maxnb) {
  __shared__ float red[256];
  int b = blockIdx.x;
  float mx = meanb[b * 3 + 0], my = meanb[b * 3 + 1], mz = meanb[b * 3 + 2];
  float m = 0.f;
  for (int p = threadIdx.x; p < NPTS; p += blockDim.x) {
    float dx = coords[((size_t)b * 3 + 0) * NPTS + p] - mx;
    float dy = coords[((size_t)b * 3 + 1) * NPTS + p] - my;
    float dz = coords[((size_t)b * 3 + 2) * NPTS + p] - mz;
    m = fmaxf(m, sqrtf(dx * dx + dy * dy + dz * dz));
  }
  red[threadIdx.x] = m; __syncthreads();
  for (int off = 128; off > 0; off >>= 1) {
    if (threadIdx.x < off) red[threadIdx.x] = fmaxf(red[threadIdx.x], red[threadIdx.x + off]);
    __syncthreads();
  }
  if (threadIdx.x == 0) maxnb[b] = red[0];
}

__global__ void compute_points_kernel(const float* __restrict__ coords,
                                      const float* __restrict__ meanb,
                                      const float* __restrict__ maxnb,
                                      float* __restrict__ ncb, float* __restrict__ w3,
                                      int* __restrict__ idx3, float* __restrict__ outc) {
  int p = blockIdx.x * blockDim.x + threadIdx.x;
  if (p >= NTOT) return;
  int b = p / NPTS, pp = p - b * NPTS;
  float scl = 1.f / (maxnb[b] * 2.f + EPSV);
  float dd[3]; int ci[3];
  #pragma unroll
  for (int a = 0; a < 3; ++a) {
    float c = coords[((size_t)b * 3 + a) * NPTS + pp];
    outc[((size_t)b * 3 + a) * NPTS + pp] = c;               // second tuple output
    float v = (c - meanb[b * 3 + a]) * scl + 0.5f;
    v = v * (RES - 1);
    v = fminf(fmaxf(v, 0.f), (float)(RES - 1) - EPSV);
    ncb[((size_t)b * 3 + a) * NPTS + pp] = v;
    float r = rintf(v);
    ci[a] = (int)r;
    dd[a] = fabsf(v - r);
  }
  #pragma unroll
  for (int a = 0; a < 3; ++a) {
    int i = (a == 0) ? 1 : 0;
    int j = (a == 2) ? 1 : 2;
    w3[p * 3 + a]  = (1.f - dd[i]) * (1.f - dd[j]);
    idx3[p * 3 + a] = (b * RES + ci[i]) * RES + ci[j];
  }
}

// ---------------------------------------------------------------------------
// WMMA GEMM: input in (B, 64, NP) feature layout (proj & point MLPs)
// wave computes 16 points x 64 out-channels; writes pre-BN activations,
// accumulates per-channel sum / sumsq for batch statistics.
// ---------------------------------------------------------------------------
__global__ void gemm_feat_kernel(const float* __restrict__ x, const _Float16* __restrict__ w,
                                 const float* __restrict__ bias, float* __restrict__ y,
                                 float* __restrict__ stat) {
  const int lane  = threadIdx.x & 31;
  const int wv    = (blockIdx.x * blockDim.x + threadIdx.x) >> 5;
  const int pbase = wv * 16;
  const int b  = pbase / NPTS;
  const int pp = pbase - b * NPTS;
  const int nli = lane & 15;
  const int hi  = lane >> 4;
  const float* xb = x + ((size_t)b * MIDC) * NPTS + pp + nli;
  v8f acc[4] = { v8f{}, v8f{}, v8f{}, v8f{} };
  #pragma unroll
  for (int kb = 0; kb < 64; kb += 32) {
    v16h bv;
    #pragma unroll
    for (int i = 0; i < 16; ++i)
      bv[i] = (_Float16)xb[(size_t)(kb + hi * 16 + i) * NPTS];
    #pragma unroll
    for (int t = 0; t < 4; ++t) {
      const int oc = t * 16 + nli;
      const _Float16* wr = w + oc * 64 + kb + hi * 16;
      v16h av;
      #pragma unroll
      for (int i = 0; i < 16; ++i) av[i] = wr[i];
      acc[t] = __builtin_amdgcn_wmma_f32_16x16x32_f16(false, av, false, bv,
                                                      (short)0, acc[t], false, false);
    }
  }
  float* yr = y + (size_t)(pbase + nli) * 64;
  #pragma unroll
  for (int t = 0; t < 4; ++t) {
    #pragma unroll
    for (int r = 0; r < 8; ++r) {
      const int m = t * 16 + hi * 8 + r;
      float v = acc[t][r] + bias[m];
      yr[m] = v;
      float s = wred16(v);
      float q = wred16(v * v);
      if (nli == 0) { atomicAdd(&stat[m], s); atomicAdd(&stat[128 + m], q); }
    }
  }
}

// ---------------------------------------------------------------------------
// WMMA GEMM: input point-major [NTOT][64], optional per-channel BN scale/shift
// (+ReLU) applied on load, optional per-batch additive vector (att2a gmax term).
// ---------------------------------------------------------------------------
__global__ void gemm_pm_kernel(const float* __restrict__ x, const float* __restrict__ instat,
                               int act, const _Float16* __restrict__ w,
                               const float* __restrict__ bias, const float* __restrict__ addvec,
                               float* __restrict__ y, float* __restrict__ stat, int ocoff) {
  const int lane  = threadIdx.x & 31;
  const int wv    = (blockIdx.x * blockDim.x + threadIdx.x) >> 5;
  const int pbase = wv * 16;
  const int nli = lane & 15;
  const int hi  = lane >> 4;
  const float* xr = x + (size_t)(pbase + nli) * 64;
  const float* avp = addvec ? (addvec + (pbase / NPTS) * 64) : (const float*)nullptr;
  v8f acc[4] = { v8f{}, v8f{}, v8f{}, v8f{} };
  #pragma unroll
  for (int kb = 0; kb < 64; kb += 32) {
    v16h bv;
    #pragma unroll
    for (int i = 0; i < 16; ++i) {
      const int c = kb + hi * 16 + i;
      float v = xr[c];
      if (instat) {
        v = instat[256 + c] * v + instat[384 + c];
        if (act) v = fmaxf(v, 0.f);
      }
      bv[i] = (_Float16)v;
    }
    #pragma unroll
    for (int t = 0; t < 4; ++t) {
      const int oc = t * 16 + nli;
      const _Float16* wr = w + oc * 64 + kb + hi * 16;
      v16h av;
      #pragma unroll
      for (int i = 0; i < 16; ++i) av[i] = wr[i];
      acc[t] = __builtin_amdgcn_wmma_f32_16x16x32_f16(false, av, false, bv,
                                                      (short)0, acc[t], false, false);
    }
  }
  float* yr = y + (size_t)(pbase + nli) * 64;
  #pragma unroll
  for (int t = 0; t < 4; ++t) {
    #pragma unroll
    for (int r = 0; r < 8; ++r) {
      const int m = t * 16 + hi * 8 + r;
      float v = acc[t][r] + bias[m] + (avp ? avp[m] : 0.f);
      yr[m] = v;
      float s = wred16(v);
      float q = wred16(v * v);
      if (nli == 0) { atomicAdd(&stat[ocoff + m], s); atomicAdd(&stat[128 + ocoff + m], q); }
    }
  }
}

// ---------------------------------------------------------------------------
// 3x3 SAME conv 64->64 as implicit WMMA GEMM over grid [NCELL][64].
// wave = 16 pixels (one x-tile of a row) x 64 output channels.
// instat != null => input is pre-BN: apply scale/shift + LeakyReLU(0.1) on load.
// ---------------------------------------------------------------------------
__global__ void conv3x3_kernel(const float* __restrict__ gin, const float* __restrict__ instat,
                               const _Float16* __restrict__ w, const float* __restrict__ bias,
                               float* __restrict__ gout, float* __restrict__ stat) {
  const int lane = threadIdx.x & 31;
  const int wv   = (blockIdx.x * blockDim.x + threadIdx.x) >> 5;   // 0..2047
  const int b    = wv >> 8;                  // 4 tiles * 64 rows per batch
  const int rem  = wv & 255;
  const int yy0  = rem >> 2;
  const int xb   = (rem & 3) * 16;
  const int nli = lane & 15;
  const int hi  = lane >> 4;
  v8f acc[4] = { v8f{}, v8f{}, v8f{}, v8f{} };
  for (int dy = 0; dy < 3; ++dy) {
    const int yy = yy0 + dy - 1;
    if (yy < 0 || yy >= RES) continue;                  // uniform per wave
    for (int dx = 0; dx < 3; ++dx) {
      const int xx  = xb + nli + dx - 1;
      const bool inb = (xx >= 0) && (xx < RES);
      const int xxc = inb ? xx : 0;
      const float* cell = gin + (size_t)((b * RES + yy) * RES + xxc) * 64;
      const int tap = dy * 3 + dx;
      #pragma unroll
      for (int kc = 0; kc < 2; ++kc) {
        v16h bv;
        #pragma unroll
        for (int i = 0; i < 16; ++i) {
          const int c = kc * 32 + hi * 16 + i;
          float v = inb ? cell[c] : 0.f;
          if (instat && inb) {
            v = instat[256 + c] * v + instat[384 + c];
            v = v > 0.f ? v : 0.1f * v;                 // LeakyReLU(0.1)
          }
          bv[i] = (_Float16)v;
        }
        #pragma unroll
        for (int t = 0; t < 4; ++t) {
          const int oc = t * 16 + nli;
          const _Float16* wr = w + oc * 576 + tap * 64 + kc * 32 + hi * 16;
          v16h av;
          #pragma unroll
          for (int i = 0; i < 16; ++i) av[i] = wr[i];
          acc[t] = __builtin_amdgcn_wmma_f32_16x16x32_f16(false, av, false, bv,
                                                          (short)0, acc[t], false, false);
        }
      }
    }
  }
  float* yr = gout + (size_t)((b * RES + yy0) * RES + xb + nli) * 64;
  #pragma unroll
  for (int t = 0; t < 4; ++t) {
    #pragma unroll
    for (int r = 0; r < 8; ++r) {
      const int m = t * 16 + hi * 8 + r;
      float v = acc[t][r] + bias[m];
      yr[m] = v;
      float s = wred16(v);
      float q = wred16(v * v);
      if (nli == 0) { atomicAdd(&stat[m], s); atomicAdd(&stat[128 + m], q); }
    }
  }
}

// ---------------------------------------------------------------------------
// attention chain (small-K pieces in VALU)
// ---------------------------------------------------------------------------
__global__ void att1a_kernel(const float* __restrict__ coords, const float* __restrict__ ncb,
                             const float* __restrict__ w, const float* __restrict__ bias,
                             float* __restrict__ y, float* __restrict__ stat) {
  __shared__ float ss[64], sq[64];
  if (threadIdx.x < 64) { ss[threadIdx.x] = 0.f; sq[threadIdx.x] = 0.f; }
  __syncthreads();
  int p = blockIdx.x * blockDim.x + threadIdx.x;
  int b = p / NPTS, pp = p - b * NPTS;
  float xin[6];
  #pragma unroll
  for (int k = 0; k < 3; ++k) xin[k]     = coords[((size_t)b * 3 + k) * NPTS + pp];
  #pragma unroll
  for (int k = 0; k < 3; ++k) xin[3 + k] = ncb[((size_t)b * 3 + k) * NPTS + pp];
  float* yr = y + (size_t)p * 64;
  for (int oc = 0; oc < 64; ++oc) {
    float a = bias[oc];
    #pragma unroll
    for (int k = 0; k < 6; ++k) a += w[oc * 6 + k] * xin[k];
    yr[oc] = a;
    atomicAdd(&ss[oc], a);
    atomicAdd(&sq[oc], a * a);
  }
  __syncthreads();
  if (threadIdx.x < 64) {
    atomicAdd(&stat[threadIdx.x], ss[threadIdx.x]);
    atomicAdd(&stat[128 + threadIdx.x], sq[threadIdx.x]);
  }
}

__global__ void gmax_kernel(const float* __restrict__ f1b, const float* __restrict__ stat3,
                            float* __restrict__ gmaxb) {
  __shared__ float red[256];
  int b = blockIdx.x >> 6, ch = blockIdx.x & 63;
  float sc = stat3[256 + ch], sh = stat3[384 + ch];
  float m = -3.4e38f;
  for (int pp = threadIdx.x; pp < NPTS; pp += blockDim.x) {
    float v = sc * f1b[(size_t)(b * NPTS + pp) * 64 + ch] + sh;
    m = fmaxf(m, v);
  }
  red[threadIdx.x] = m; __syncthreads();
  for (int off = 128; off > 0; off >>= 1) {
    if (threadIdx.x < off) red[threadIdx.x] = fmaxf(red[threadIdx.x], red[threadIdx.x + off]);
    __syncthreads();
  }
  if (threadIdx.x == 0) gmaxb[b * 64 + ch] = red[0];
}

__global__ void gconst_kernel(const float* __restrict__ w2a, const float* __restrict__ gmaxb,
                              float* __restrict__ gconstb) {
  int i = blockIdx.x * blockDim.x + threadIdx.x;     // 512
  if (i >= NBATCH * 64) return;
  int b = i >> 6, oc = i & 63;
  float a = 0.f;
  for (int c = 0; c < 64; ++c) a += w2a[oc * 128 + 64 + c] * gmaxb[b * 64 + c];
  gconstb[i] = a;
}

__global__ void att2b_kernel(const float* __restrict__ a2, const float* __restrict__ stat4,
                             const float* __restrict__ w, const float* __restrict__ bias,
                             float* __restrict__ att4, float* __restrict__ stat5) {
  __shared__ float ss[4], sq[4];
  if (threadIdx.x < 4) { ss[threadIdx.x] = 0.f; sq[threadIdx.x] = 0.f; }
  __syncthreads();
  int p = blockIdx.x * blockDim.x + threadIdx.x;
  float y[4];
  #pragma unroll
  for (int k = 0; k < 4; ++k) y[k] = bias[k];
  const float* xr = a2 + (size_t)p * 64;
  for (int c = 0; c < 64; ++c) {
    float v = fmaxf(stat4[256 + c] * xr[c] + stat4[384 + c], 0.f);
    #pragma unroll
    for (int k = 0; k < 4; ++k) y[k] += w[k * 64 + c] * v;
  }
  #pragma unroll
  for (int k = 0; k < 4; ++k) {
    att4[(size_t)p * 4 + k] = y[k];
    atomicAdd(&ss[k], y[k]);
    atomicAdd(&sq[k], y[k] * y[k]);
  }
  __syncthreads();
  if (threadIdx.x < 4) {
    atomicAdd(&stat5[threadIdx.x], ss[threadIdx.x]);
    atomicAdd(&stat5[128 + threadIdx.x], sq[threadIdx.x]);
  }
}

__global__ void softmax_kernel(float* att4, const float* __restrict__ stat5) {
  int p = blockIdx.x * blockDim.x + threadIdx.x;
  float y[4], m = -3.4e38f;
  #pragma unroll
  for (int k = 0; k < 4; ++k) {
    y[k] = stat5[256 + k] * att4[(size_t)p * 4 + k] + stat5[384 + k];
    m = fmaxf(m, y[k]);
  }
  float s = 0.f;
  #pragma unroll
  for (int k = 0; k < 4; ++k) { y[k] = expf(y[k] - m); s += y[k]; }
  float inv = 1.f / s;
  #pragma unroll
  for (int k = 0; k < 4; ++k) att4[(size_t)p * 4 + k] = y[k] * inv;
}

// ---------------------------------------------------------------------------
// fusion, scatter / gather
// ---------------------------------------------------------------------------
__global__ void init_fused_kernel(float* fused /* ptfeat, in place */,
                                  const float* __restrict__ att4,
                                  const float* __restrict__ stat1) {
  int g = blockIdx.x * blockDim.x + threadIdx.x;     // NTOT*64
  int p = g >> 6, c = g & 63;
  float v = fmaxf(stat1[256 + c] * fused[g] + stat1[384 + c], 0.f);
  fused[g] = att4[(size_t)p * 4 + 0] * v;
}

__global__ void scatter_kernel(const float* __restrict__ pfeat, const float* __restrict__ stat0,
                               const float* __restrict__ w3, const int* __restrict__ idx3,
                               int a, float* __restrict__ num, float* __restrict__ den) {
  int g = blockIdx.x * blockDim.x + threadIdx.x;     // NTOT*64
  int p = g >> 6, c = g & 63;
  float v = fmaxf(stat0[256 + c] * pfeat[g] + stat0[384 + c], 0.f);
  float wgt = w3[p * 3 + a];
  int idx = idx3[p * 3 + a];
  atomicAdd(&num[(size_t)idx * 64 + c], v * wgt);
  if (c == 0) atomicAdd(&den[idx], wgt);
}

__global__ void normalize_grid_kernel(float* num, const float* __restrict__ den) {
  int g = blockIdx.x * blockDim.x + threadIdx.x;     // NCELL*64
  num[g] = num[g] / (den[g >> 6] + EPSV);
}

__global__ void gather_accum_kernel(float* __restrict__ fused, const float* __restrict__ grid0,
                                    const float* __restrict__ grid2, const float* __restrict__ statc2,
                                    const float* __restrict__ att4, const float* __restrict__ w3,
                                    const int* __restrict__ idx3, int a) {
  int g = blockIdx.x * blockDim.x + threadIdx.x;     // NTOT*64
  int p = g >> 6, c = g & 63;
  int idx = idx3[p * 3 + a];
  float wgt = w3[p * 3 + a];
  float pre = statc2[256 + c] * grid2[(size_t)idx * 64 + c] + statc2[384 + c];
  float y2  = pre > 0.f ? pre : 0.1f * pre;           // LeakyReLU(0.1)
  float val = grid0[(size_t)idx * 64 + c] + y2;       // residual
  fused[g] += att4[(size_t)p * 4 + 1 + a] * wgt * val;
}

__global__ void final_out_kernel(const float* __restrict__ lastA, const float* __restrict__ lastB,
                                 const float* __restrict__ stat6, float* __restrict__ out) {
  int g = blockIdx.x * blockDim.x + threadIdx.x;     // NTOT*128
  int p = g >> 7, oc = g & 127;
  int b = p / NPTS, pp = p - b * NPTS;
  float pre = (oc < 64) ? lastA[(size_t)p * 64 + oc] : lastB[(size_t)p * 64 + (oc - 64)];
  float v = fmaxf(stat6[256 + oc] * pre + stat6[384 + oc], 0.f);
  out[((size_t)b * OUTC + oc) * NPTS + pp] = v;
}

// ---------------------------------------------------------------------------
// host launcher
// ---------------------------------------------------------------------------
extern "C" void kernel_launch(void* const* d_in, const int* in_sizes, int n_in,
                              void* d_out, int out_size, void* d_ws, size_t ws_size,
                              hipStream_t stream) {
  (void)n_in; (void)out_size; (void)ws_size;
  // features/coords order (disambiguate via size); param leaves follow at [2..]
  int fi = 0, cix = 1;
  if (in_sizes[0] == NBATCH * 3 * NPTS) { fi = 1; cix = 0; }
  const float* features = (const float*)d_in[fi];
  const float* coords   = (const float*)d_in[cix];
  // sorted pytree leaves of params:
  // att1a{b,beta,g,w}=0..3  att1b=4..7  att2a=8..11  att2b=12..15
  // last=16..19  point=20..23  proj=24..27  x=28..35  y=36..43  z=44..51
  // plane leaves: b1+0 b2+1 be1+2 be2+3 g1+4 g2+5 w1+6 w2+7
  #define PIN(i) ((const float*)d_in[2 + (i)])

  // -------- workspace carve --------
  size_t cur = 0;
  char* base = (char*)d_ws;
  auto alloc = [&](size_t bytes) -> void* {
    void* p = base + cur;
    cur = (cur + bytes + 255) & ~(size_t)255;
    return p;
  };
  float*    stats   = (float*)alloc(16 * 512 * sizeof(float));
  _Float16* wproj   = (_Float16*)alloc(4096 * 2);
  _Float16* wpoint  = (_Float16*)alloc(4096 * 2);
  _Float16* watt1b  = (_Float16*)alloc(4096 * 2);
  _Float16* watt2a  = (_Float16*)alloc(4096 * 2);
  _Float16* wlast   = (_Float16*)alloc(8192 * 2);
  _Float16* wconv[6];
  for (int i = 0; i < 6; ++i) wconv[i] = (_Float16*)alloc(36864 * 2);
  float* meanb   = (float*)alloc(24 * 4);
  float* maxnb   = (float*)alloc(8 * 4);
  float* gmaxb   = (float*)alloc(512 * 4);
  float* gconstb = (float*)alloc(512 * 4);
  float* ncb   = (float*)alloc((size_t)NTOT * 3 * 4);
  float* w3    = (float*)alloc((size_t)NTOT * 3 * 4);
  int*   idx3  = (int*)  alloc((size_t)NTOT * 3 * 4);
  float* pfeat  = (float*)alloc((size_t)NTOT * 64 * 4);   // proj pre-act; later lastA
  float* ptfeat = (float*)alloc((size_t)NTOT * 64 * 4);   // point pre-act; later fused
  float* f1a    = (float*)alloc((size_t)NTOT * 64 * 4);   // att1a pre; later att2a pre (a2)
  float* f1b    = (float*)alloc((size_t)NTOT * 64 * 4);   // att1b pre; later lastB
  float* att4   = (float*)alloc((size_t)NTOT * 4 * 4);
  float* gridnum = (float*)alloc((size_t)(NCELL * 64 + NCELL) * 4); // num + den contiguous
  float* gridden = gridnum + (size_t)NCELL * 64;
  float* grid1   = (float*)alloc((size_t)NCELL * 64 * 4);
  float* grid2   = (float*)alloc((size_t)NCELL * 64 * 4);

  float* outc = (float*)d_out + (size_t)NBATCH * OUTC * NPTS;  // coords tuple slot
  auto statL = [&](int i) { return stats + i * 512; };
  const float invNT = 1.f / (float)NTOT;
  const float invNC = 1.f / (float)NCELL;

  // -------- stats zero + weight conversion --------
  zero_kernel<<<32, 256, 0, stream>>>(stats, 16 * 512);
  cvt_f16_kernel<<<16, 256, 0, stream>>>(PIN(27), wproj, 4096);
  cvt_f16_kernel<<<16, 256, 0, stream>>>(PIN(23), wpoint, 4096);
  cvt_f16_kernel<<<16, 256, 0, stream>>>(PIN(7),  watt1b, 4096);
  cvt_att2a_kernel<<<16, 256, 0, stream>>>(PIN(11), watt2a);
  cvt_f16_kernel<<<32, 256, 0, stream>>>(PIN(19), wlast, 8192);
  const int pbi[3] = {28, 36, 44};
  for (int a = 0; a < 3; ++a) {
    cvt_convw_kernel<<<144, 256, 0, stream>>>(PIN(pbi[a] + 6), wconv[a * 2 + 0]);
    cvt_convw_kernel<<<144, 256, 0, stream>>>(PIN(pbi[a] + 7), wconv[a * 2 + 1]);
  }

  // -------- coordinates --------
  batch_mean_kernel<<<NBATCH, 256, 0, stream>>>(coords, meanb);
  batch_maxnorm_kernel<<<NBATCH, 256, 0, stream>>>(coords, meanb, maxnb);
  compute_points_kernel<<<NTOT / 256, 256, 0, stream>>>(coords, meanb, maxnb,
                                                        ncb, w3, idx3, outc);

  // -------- proj / point MLPs (WMMA) --------
  gemm_feat_kernel<<<NTOT / (16 * 4), 128, 0, stream>>>(features, wproj, PIN(24), pfeat, statL(0));
  gemm_feat_kernel<<<NTOT / (16 * 4), 128, 0, stream>>>(features, wpoint, PIN(20), ptfeat, statL(1));
  att1a_kernel<<<NTOT / 256, 256, 0, stream>>>(coords, ncb, PIN(3), PIN(0), f1a, statL(2));
  finalize_bn_kernel<<<1, 128, 0, stream>>>(statL(0), PIN(26), PIN(25), 64, invNT);
  finalize_bn_kernel<<<1, 128, 0, stream>>>(statL(1), PIN(22), PIN(21), 64, invNT);
  finalize_bn_kernel<<<1, 128, 0, stream>>>(statL(2), PIN(2),  PIN(1),  64, invNT);

  // -------- attention chain --------
  gemm_pm_kernel<<<NTOT / 64, 128, 0, stream>>>(f1a, statL(2), 1, watt1b, PIN(4),
                                                nullptr, f1b, statL(3), 0);
  finalize_bn_kernel<<<1, 128, 0, stream>>>(statL(3), PIN(6), PIN(5), 64, invNT);
  gmax_kernel<<<NBATCH * 64, 256, 0, stream>>>(f1b, statL(3), gmaxb);
  gconst_kernel<<<2, 256, 0, stream>>>(PIN(11), gmaxb, gconstb);
  gemm_pm_kernel<<<NTOT / 64, 128, 0, stream>>>(f1b, statL(3), 0, watt2a, PIN(8),
                                                gconstb, f1a, statL(4), 0);
  finalize_bn_kernel<<<1, 128, 0, stream>>>(statL(4), PIN(10), PIN(9), 64, invNT);
  att2b_kernel<<<NTOT / 256, 256, 0, stream>>>(f1a, statL(4), PIN(15), PIN(12), att4, statL(5));
  finalize_bn_kernel<<<1, 128, 0, stream>>>(statL(5), PIN(14), PIN(13), 4, invNT);
  softmax_kernel<<<NTOT / 256, 256, 0, stream>>>(att4, statL(5));

  // -------- fused accumulation: att0 * point_feat --------
  init_fused_kernel<<<NTOT * 64 / 256, 256, 0, stream>>>(ptfeat, att4, statL(1));

  // -------- per-plane scatter -> conv stack (WMMA) -> gather --------
  for (int a = 0; a < 3; ++a) {
    const int pb = pbi[a];
    zero_kernel<<<(NCELL * 64 + NCELL + 255) / 256, 256, 0, stream>>>(gridnum, NCELL * 64 + NCELL);
    scatter_kernel<<<NTOT * 64 / 256, 256, 0, stream>>>(pfeat, statL(0), w3, idx3, a,
                                                        gridnum, gridden);
    normalize_grid_kernel<<<NCELL * 64 / 256, 256, 0, stream>>>(gridnum, gridden);
    conv3x3_kernel<<<512, 128, 0, stream>>>(gridnum, nullptr, wconv[a * 2 + 0],
                                            PIN(pb + 0), grid1, statL(7 + a * 2));
    finalize_bn_kernel<<<1, 128, 0, stream>>>(statL(7 + a * 2), PIN(pb + 4), PIN(pb + 2), 64, invNC);
    conv3x3_kernel<<<512, 128, 0, stream>>>(grid1, statL(7 + a * 2), wconv[a * 2 + 1],
                                            PIN(pb + 1), grid2, statL(8 + a * 2));
    finalize_bn_kernel<<<1, 128, 0, stream>>>(statL(8 + a * 2), PIN(pb + 5), PIN(pb + 3), 64, invNC);
    gather_accum_kernel<<<NTOT * 64 / 256, 256, 0, stream>>>(ptfeat, gridnum, grid2,
                                                             statL(8 + a * 2), att4, w3, idx3, a);
  }

  // -------- last MLP (64 -> 128) as two WMMA GEMMs --------
  gemm_pm_kernel<<<NTOT / 64, 128, 0, stream>>>(ptfeat, nullptr, 0, wlast, PIN(16),
                                                nullptr, pfeat, statL(6), 0);
  gemm_pm_kernel<<<NTOT / 64, 128, 0, stream>>>(ptfeat, nullptr, 0, wlast + 64 * 64, PIN(16) + 64,
                                                nullptr, f1b, statL(6), 64);
  finalize_bn_kernel<<<1, 128, 0, stream>>>(statL(6), PIN(18), PIN(17), 128, invNT);
  final_out_kernel<<<NTOT * 128 / 256, 256, 0, stream>>>(pfeat, f1b, statL(6), (float*)d_out);
  #undef PIN
}
